// OrientedRCNNHead_80865644249697
// MI455X (gfx1250) — matrix operand order repaired and checked by reading, not compiled, not run on hardware
//
#include <hip/hip_runtime.h>
#include <cstdint>
#include <cstddef>

// Problem constants (from reference)
#define B_    2
#define N_    512
#define C_    256
#define H_    256
#define W_    256
#define OUT_  1024
#define K1_   12544      // C*7*7
#define M_    1024       // B*N
#define NCLS_ 11

typedef __bf16 bfv16 __attribute__((ext_vector_type(16)));
typedef __bf16 bfv8  __attribute__((ext_vector_type(8)));
typedef float  fv8   __attribute__((ext_vector_type(8)));
typedef int    v4i   __attribute__((ext_vector_type(4)));
typedef __attribute__((address_space(1))) v4i* g4ptr;   // global int4*
typedef __attribute__((address_space(3))) v4i* l4ptr;   // LDS int4*

// CDNA5 async global->LDS path (ASYNCcnt), if the toolchain declares it.
#if defined(__has_builtin)
#if __has_builtin(__builtin_amdgcn_global_load_async_to_lds_b128) && \
    __has_builtin(__builtin_amdgcn_s_wait_asynccnt)
#define USE_ASYNC_LDS 1
#endif
#endif
#ifndef USE_ASYNC_LDS
#define USE_ASYNC_LDS 0
#endif

__device__ __forceinline__ unsigned short f2bf(float f) {
  unsigned u = __builtin_bit_cast(unsigned, f);
  u += 0x7FFFu + ((u >> 16) & 1u);          // round-to-nearest-even
  return (unsigned short)(u >> 16);
}
__device__ __forceinline__ float bf2f(unsigned short h) {
  unsigned u = ((unsigned)h) << 16;
  return __builtin_bit_cast(float, u);
}

// ---------------------------------------------------------------------------
// Kernel 1: feat NCHW f32 -> NHWC bf16 (coalesced both sides via LDS tile)
// grid (W/32, C/32, B*H), block (32,8)
// ---------------------------------------------------------------------------
__global__ __launch_bounds__(256) void nchw_to_nhwc(
    const float* __restrict__ in, unsigned short* __restrict__ out) {
  __shared__ float tile[32][33];
  const int wt = blockIdx.x * 32, ct = blockIdx.y * 32;
  const int b = blockIdx.z >> 8, h = blockIdx.z & 255;
  const int tx = threadIdx.x, ty = threadIdx.y;
#pragma unroll
  for (int i = 0; i < 4; ++i) {
    int c = ct + ty + i * 8;
    tile[ty + i * 8][tx] =
        in[(((size_t)b * C_ + c) * H_ + h) * W_ + wt + tx];
  }
  __syncthreads();
#pragma unroll
  for (int i = 0; i < 4; ++i) {
    int w = wt + ty + i * 8;
    out[(((size_t)b * H_ + h) * W_ + w) * C_ + ct + tx] =
        f2bf(tile[tx][ty + i * 8]);
  }
}

// ---------------------------------------------------------------------------
// Kernel 2: W (K x N, f32) -> WT (N x K, bf16), optional row permutation so
// GEMM's K index is (bin*256 + c) instead of (c*49 + bin).
// grid (K/32, N/32), block (32,8)
// ---------------------------------------------------------------------------
template <bool PERM>
__global__ __launch_bounds__(256) void w_transpose(
    const float* __restrict__ Wm, unsigned short* __restrict__ WT,
    int K, int N) {
  __shared__ float tile[32][33];
  const int kt = blockIdx.x * 32, nt = blockIdx.y * 32;
  const int tx = threadIdx.x, ty = threadIdx.y;
#pragma unroll
  for (int i = 0; i < 4; ++i) {
    int kn = kt + ty + i * 8;                       // new K index
    int ko = PERM ? ((kn & 255) * 49 + (kn >> 8))   // c*49 + bin
                  : kn;
    tile[ty + i * 8][tx] = Wm[(size_t)ko * N + nt + tx];
  }
  __syncthreads();
#pragma unroll
  for (int i = 0; i < 4; ++i) {
    int n = nt + ty + i * 8;
    WT[(size_t)n * K + kt + tx] = f2bf(tile[tx][ty + i * 8]);
  }
}

// ---------------------------------------------------------------------------
// Kernel 3: rotated ROI-align, one block per proposal, 256 threads = channels.
// Writes X[row][bin*256 + c] bf16 (coalesced 512B per bin store).
// ---------------------------------------------------------------------------
__global__ __launch_bounds__(256) void roi_pool(
    const unsigned short* __restrict__ feat_hwc,   // [B][H][W][C] bf16
    const float* __restrict__ prop,                // [B*N][5]
    unsigned short* __restrict__ X) {              // [M][K1]
  const int bn = blockIdx.x;
  const int b  = bn >> 9;                           // N_ = 512
  const float* p = prop + (size_t)bn * 5;

  __shared__ float s_ly[196], s_lx[196], s_sc[196];
  __shared__ int   s_o[196][4];

  const int t = threadIdx.x;
  if (t < 196) {
    float cx = p[0], cy = p[1], w = p[2], h = p[3], th = p[4];
    int sx = t & 1, sy = (t >> 1) & 1, bin = t >> 2;
    int px = bin % 7, py = bin / 7;
    float ys = -h * 0.5f + (h / 7.0f) * ((float)py + (sy + 0.5f) * 0.5f);
    float xs = -w * 0.5f + (w / 7.0f) * ((float)px + (sx + 0.5f) * 0.5f);
    float ct = cosf(th), st = sinf(th);
    float gx = cx + xs * ct - ys * st;
    float gy = cy + xs * st + ys * ct;
    bool valid = (gy > -1.0f) && (gy < 256.0f) && (gx > -1.0f) && (gx < 256.0f);
    float y = fminf(fmaxf(gy, 0.0f), 255.0f);
    float x = fminf(fmaxf(gx, 0.0f), 255.0f);
    float y0f = floorf(y), x0f = floorf(x);
    int y0 = (int)y0f, x0 = (int)x0f;
    int y1 = min(y0 + 1, 255), x1 = min(x0 + 1, 255);
    s_ly[t] = y - y0f;
    s_lx[t] = x - x0f;
    s_sc[t] = valid ? 0.25f : 0.0f;                 // fold in mean over 2x2 subs
    s_o[t][0] = (y0 * 256 + x0) * 256;
    s_o[t][1] = (y0 * 256 + x1) * 256;
    s_o[t][2] = (y1 * 256 + x0) * 256;
    s_o[t][3] = (y1 * 256 + x1) * 256;
  }
  __syncthreads();

  const unsigned short* fb = feat_hwc + (size_t)b * H_ * W_ * C_;
  unsigned short* xrow = X + (size_t)bn * K1_;
  const int c = t;
  for (int bin = 0; bin < 49; ++bin) {
    float acc = 0.0f;
#pragma unroll
    for (int s4 = 0; s4 < 4; ++s4) {
      int si = bin * 4 + s4;
      float sc = s_sc[si], ly = s_ly[si], lx = s_lx[si];
      float hy = 1.0f - ly, hx = 1.0f - lx;
      float v00 = bf2f(fb[s_o[si][0] + c]);
      float v01 = bf2f(fb[s_o[si][1] + c]);
      float v10 = bf2f(fb[s_o[si][2] + c]);
      float v11 = bf2f(fb[s_o[si][3] + c]);
      acc += sc * (hy * (hx * v00 + lx * v01) + ly * (hx * v10 + lx * v11));
    }
    xrow[bin * 256 + c] = f2bf(acc);
  }
}

// ---------------------------------------------------------------------------
// Kernel 4: bf16 WMMA GEMM  D = relu(A(MxK) * B(KxN) + bias), B given as NxK.
// Block tile 64(M) x 128(N), 8 waves (2x4), wave tile 32x32, k-step 32,
// double-buffered LDS with 40-element row padding (conflict-free b128 reads).
// Staging uses GLOBAL_LOAD_ASYNC_TO_LDS_B128 (ASYNCcnt) when available.
// ---------------------------------------------------------------------------
#define LDT 40
template <bool OUT_BF16>
__global__ __launch_bounds__(256) void gemm_wmma(
    const unsigned short* __restrict__ A,    // M x K bf16
    const unsigned short* __restrict__ Bt,   // N x K bf16
    const float* __restrict__ bias,          // N
    void* __restrict__ outp,                 // M x N
    int Mdim, int Ndim, int Kdim) {
  __shared__ __attribute__((aligned(16))) unsigned short sA[2][64 * LDT];
  __shared__ __attribute__((aligned(16))) unsigned short sB[2][128 * LDT];

  const int tid  = threadIdx.x;
  const int lane = tid & 31;
  const int wv   = tid >> 5;
  const int wm   = wv & 1;       // 2 waves in M
  const int wn   = wv >> 1;      // 4 waves in N
  const int ln   = lane & 15;
  const int hh   = lane >> 4;
  const int m0   = blockIdx.y * 64;
  const int n0   = blockIdx.x * 128;
  const int KT   = Kdim >> 5;

  const int arow  = tid >> 2, ac = tid & 3;   // A: 256 16B chunks / stage
  const int brow0 = tid >> 2, brow1 = brow0 + 64;

#if USE_ASYNC_LDS
  // Direct global -> LDS (no VGPR staging), tracked with ASYNCcnt.
  auto stage_async = [&](int kk, int buf) {
    const size_t ko = (size_t)kk * 32;
    __builtin_amdgcn_global_load_async_to_lds_b128(
        (g4ptr)(A + (size_t)(m0 + arow) * Kdim + ko + ac * 8),
        (l4ptr)(&sA[buf][arow * LDT + ac * 8]),
        0, 0);
    __builtin_amdgcn_global_load_async_to_lds_b128(
        (g4ptr)(Bt + (size_t)(n0 + brow0) * Kdim + ko + ac * 8),
        (l4ptr)(&sB[buf][brow0 * LDT + ac * 8]),
        0, 0);
    __builtin_amdgcn_global_load_async_to_lds_b128(
        (g4ptr)(Bt + (size_t)(n0 + brow1) * Kdim + ko + ac * 8),
        (l4ptr)(&sB[buf][brow1 * LDT + ac * 8]),
        0, 0);
  };
#else
  uint4 ra, rb0, rb1;
  auto gload = [&](int kk) {
    const size_t ko = (size_t)kk * 32;
    ra  = *(const uint4*)(A  + (size_t)(m0 + arow)  * Kdim + ko + ac * 8);
    rb0 = *(const uint4*)(Bt + (size_t)(n0 + brow0) * Kdim + ko + ac * 8);
    rb1 = *(const uint4*)(Bt + (size_t)(n0 + brow1) * Kdim + ko + ac * 8);
  };
  auto lstore = [&](int buf) {
    *(uint4*)(&sA[buf][arow  * LDT + ac * 8]) = ra;
    *(uint4*)(&sB[buf][brow0 * LDT + ac * 8]) = rb0;
    *(uint4*)(&sB[buf][brow1 * LDT + ac * 8]) = rb1;
  };
#endif

  fv8 acc[2][2] = {};

#if USE_ASYNC_LDS
  stage_async(0, 0);
  __builtin_amdgcn_s_wait_asynccnt(0);
  __syncthreads();
#else
  gload(0);
  lstore(0);
  __syncthreads();
#endif

  for (int kk = 0; kk < KT; ++kk) {
    const int cur = kk & 1;
#if USE_ASYNC_LDS
    if (kk + 1 < KT) stage_async(kk + 1, (kk + 1) & 1);
#else
    if (kk + 1 < KT) gload(kk + 1);
#endif

    bfv16 af[2], bfr[2];
#pragma unroll
    for (int i = 0; i < 2; ++i) {
      // 16-bit A 16x32 layout: lanes0-15 K = h*8+{0..7} | 16+h*8+{0..7}
      const unsigned short* pa = &sA[cur][(wm * 32 + i * 16 + ln) * LDT];
      bfv16 v;
      ((bfv8*)&v)[0] = *(const bfv8*)(pa + hh * 8);
      ((bfv8*)&v)[1] = *(const bfv8*)(pa + 16 + hh * 8);
      af[i] = v;
    }
#pragma unroll
    for (int j = 0; j < 2; ++j) {
      // 16-bit B 32x16: lane n=ln holds K = h*8+{0..7} | 16+h*8+{0..7}
      const unsigned short* pb = &sB[cur][(wn * 32 + j * 16 + ln) * LDT];
      bfv16 v;
      ((bfv8*)&v)[0] = *(const bfv8*)(pb + hh * 8);
      ((bfv8*)&v)[1] = *(const bfv8*)(pb + 16 + hh * 8);
      bfr[j] = v;
    }
#pragma unroll
    for (int i = 0; i < 2; ++i)
#pragma unroll
      for (int j = 0; j < 2; ++j)
        acc[i][j] = __builtin_amdgcn_wmma_f32_16x16x32_bf16(
            false, af[i], false, bfr[j], (short)0, acc[i][j], false, false);

    if (kk + 1 < KT) {
#if USE_ASYNC_LDS
      __builtin_amdgcn_s_wait_asynccnt(0);
      __syncthreads();
#else
      lstore((kk + 1) & 1);
      __syncthreads();
#endif
    }
  }

  // Epilogue: C/D layout — lane l: n = l&15, m = (l>>4)*8 + r
#pragma unroll
  for (int i = 0; i < 2; ++i)
#pragma unroll
    for (int j = 0; j < 2; ++j) {
      const int n = n0 + wn * 32 + j * 16 + ln;
      const float bv = bias[n];
#pragma unroll
      for (int r = 0; r < 8; ++r) {
        const int m = m0 + wm * 32 + i * 16 + hh * 8 + r;
        float v = acc[i][j][r] + bv;
        v = v > 0.0f ? v : 0.0f;
        if constexpr (OUT_BF16)
          ((unsigned short*)outp)[(size_t)m * Ndim + n] = f2bf(v);
        else
          ((float*)outp)[(size_t)m * Ndim + n] = v;
      }
    }
}

// ---------------------------------------------------------------------------
// Kernel 5: cls/reg heads + rotated-box decode. One block per row, 8 waves,
// each wave reduces two 1024-long dot products.
// ---------------------------------------------------------------------------
__global__ __launch_bounds__(256) void heads_kernel(
    const float* __restrict__ H2,       // [M][1024]
    const float* __restrict__ Wcls, const float* __restrict__ bcls,
    const float* __restrict__ Wreg, const float* __restrict__ breg,
    const float* __restrict__ prop,
    float* __restrict__ out_boxes, float* __restrict__ out_cls) {
  const int row = blockIdx.x;
  const float* hr = H2 + (size_t)row * OUT_;
  __shared__ float s_out[16];

  const int lane = threadIdx.x & 31, wv = threadIdx.x >> 5;
  const int j0 = wv, j1 = wv + 8;     // 0..10 -> cls, 11..15 -> reg 0..4
  float a0 = 0.0f, a1 = 0.0f;
  for (int k = lane; k < OUT_; k += 32) {
    float hv = hr[k];
    a0 += hv * Wcls[k * NCLS_ + j0];
    a1 += hv * (j1 < NCLS_ ? Wcls[k * NCLS_ + j1] : Wreg[k * 5 + (j1 - NCLS_)]);
  }
#pragma unroll
  for (int o = 16; o; o >>= 1) {
    a0 += __shfl_xor(a0, o, 32);
    a1 += __shfl_xor(a1, o, 32);
  }
  if (lane == 0) { s_out[j0] = a0; s_out[j1] = a1; }
  __syncthreads();

  if (threadIdx.x < NCLS_)
    out_cls[(size_t)row * NCLS_ + threadIdx.x] =
        s_out[threadIdx.x] + bcls[threadIdx.x];

  if (threadIdx.x == 0) {
    const float r0 = s_out[11] + breg[0], r1 = s_out[12] + breg[1];
    const float r2 = s_out[13] + breg[2], r3 = s_out[14] + breg[3];
    const float r4 = s_out[15] + breg[4];
    const float* p = prop + (size_t)row * 5;
    const float p0 = p[0] * 4.0f, p1 = p[1] * 4.0f;
    const float p2 = p[2] * 4.0f, p3 = p[3] * 4.0f, p4 = p[4];
    const float cv = 4.135166556742356f;            // |log(16/1000)|
    const float PI = 3.14159265358979323846f;
    float bx = p2 * r0 + p0;
    float by = p3 * r1 + p1;
    float bw = p2 * expf(fminf(fmaxf(r2, -cv), cv));
    float bh = p3 * expf(fminf(fmaxf(r3, -cv), cv));
    float a  = p4 + r4;
    float m  = fmodf(a + 0.5f * PI, PI);
    if (m < 0.0f) m += PI;
    float ba = m - 0.5f * PI;
    float* ob = out_boxes + (size_t)row * 5;
    ob[0] = bx; ob[1] = by; ob[2] = bw; ob[3] = bh; ob[4] = ba;
  }
}

// ---------------------------------------------------------------------------
extern "C" void kernel_launch(void* const* d_in, const int* in_sizes, int n_in,
                              void* d_out, int out_size, void* d_ws,
                              size_t ws_size, hipStream_t stream) {
  (void)in_sizes; (void)n_in; (void)out_size; (void)ws_size;
  const float* feat = (const float*)d_in[0];
  const float* prop = (const float*)d_in[1];
  const float* W1   = (const float*)d_in[2];
  const float* b1   = (const float*)d_in[3];
  const float* W2   = (const float*)d_in[4];
  const float* b2   = (const float*)d_in[5];
  const float* Wcls = (const float*)d_in[6];
  const float* bcls = (const float*)d_in[7];
  const float* Wreg = (const float*)d_in[8];
  const float* breg = (const float*)d_in[9];
  float* out_boxes = (float*)d_out;
  float* out_cls   = (float*)d_out + (size_t)M_ * 5;

  char* ws = (char*)d_ws;
  size_t off = 0;
  auto carve = [&](size_t bytes) -> char* {
    char* p = ws + off;
    off = (off + bytes + 255) & ~(size_t)255;
    return p;
  };
  unsigned short* featH = (unsigned short*)carve((size_t)B_ * H_ * W_ * C_ * 2);
  unsigned short* W1T   = (unsigned short*)carve((size_t)OUT_ * K1_ * 2);
  unsigned short* W2T   = (unsigned short*)carve((size_t)OUT_ * OUT_ * 2);
  unsigned short* Xbf   = (unsigned short*)carve((size_t)M_ * K1_ * 2);
  unsigned short* H1    = (unsigned short*)carve((size_t)M_ * OUT_ * 2);
  float*          H2    = (float*)carve((size_t)M_ * OUT_ * 4);

  nchw_to_nhwc<<<dim3(W_ / 32, C_ / 32, B_ * H_), dim3(32, 8), 0, stream>>>(
      feat, featH);
  w_transpose<true><<<dim3(K1_ / 32, OUT_ / 32), dim3(32, 8), 0, stream>>>(
      W1, W1T, K1_, OUT_);
  w_transpose<false><<<dim3(OUT_ / 32, OUT_ / 32), dim3(32, 8), 0, stream>>>(
      W2, W2T, OUT_, OUT_);
  roi_pool<<<M_, 256, 0, stream>>>(featH, prop, Xbf);
  gemm_wmma<true><<<dim3(OUT_ / 128, M_ / 64), 256, 0, stream>>>(
      Xbf, W1T, b1, (void*)H1, M_, OUT_, K1_);
  gemm_wmma<false><<<dim3(OUT_ / 128, M_ / 64), 256, 0, stream>>>(
      H1, W2T, b2, (void*)H2, M_, OUT_, OUT_);
  heads_kernel<<<M_, 256, 0, stream>>>(H2, Wcls, bcls, Wreg, breg, prop,
                                       out_boxes, out_cls);
}
